// FCOS_77000173683051
// MI455X (gfx1250) — matrix operand order repaired and checked by reading, not compiled, not run on hardware
//
#include <hip/hip_runtime.h>
#include <math.h>

#define NUM_CLASSES 80
#define CINCH 256
#define BATCHN 2
#define MAXCAND 5000
#define PRE_T 0.05f
#define NMS_T 0.6f
#define POST_K 100

typedef __attribute__((ext_vector_type(16))) __bf16 v16bf;
typedef __attribute__((ext_vector_type(8)))  float  v8f;
typedef __attribute__((ext_vector_type(4)))  unsigned int v4u;

union Frag { v16bf v; v4u u[2]; };

__device__ __forceinline__ float sigmoidf(float x){ return 1.0f/(1.0f + __expf(-x)); }

// round-to-nearest-even fp32 -> bf16 bits
__device__ __forceinline__ unsigned short f2bf(float f){
  unsigned int u = __builtin_bit_cast(unsigned int, f);
  u = (u + 0x7FFFu + ((u >> 16) & 1u)) >> 16;
  return (unsigned short)u;
}
__device__ __forceinline__ float bf2f(unsigned short h){
  unsigned int u = ((unsigned int)h) << 16;
  return __builtin_bit_cast(float, u);
}

// ---------------------------------------------------------------------------
// NCHW fp32 -> NHWC bf16 (channels innermost, 256 per pixel)
// ---------------------------------------------------------------------------
__global__ __launch_bounds__(256) void to_nhwc(const float* __restrict__ in,
                                               unsigned short* __restrict__ out,
                                               int HW){
  size_t idx = (size_t)blockIdx.x*256 + threadIdx.x;
  size_t total = (size_t)BATCHN*CINCH*HW;
  if (idx >= total) return;
  int c = (int)(idx & 255u);
  size_t pb = idx >> 8;              // b*HW + p
  int b = (int)(pb / HW);
  int p = (int)(pb - (size_t)b*HW);
  out[idx] = f2bf(in[((size_t)b*CINCH + c)*HW + p]);
}

// ---------------------------------------------------------------------------
// Pack conv weights [CO][256][3][3] fp32 into bf16 WMMA B-fragment layout:
//   frag f = (tap*8 + kb)*NBLK + nb, 256 dwords per frag, lane*8 + j inside.
//   dword j of lane L holds K = kb*32 + (L>>4)*16 + 2j (lo) and +1 (hi),
//   column co = nb*16 + (L&15). K order = tap-major, channel inner.
// ---------------------------------------------------------------------------
__global__ __launch_bounds__(256) void pack_weights(const float* __restrict__ w,
                                                    unsigned int* __restrict__ wp,
                                                    int CO){
  const int NBLK = CO >> 4;
  const size_t total = (size_t)9*8*NBLK*256;
  size_t idx = (size_t)blockIdx.x*256 + threadIdx.x;
  if (idx >= total) return;
  int j    = (int)(idx & 7u);
  int lane = (int)((idx >> 3) & 31u);
  size_t f = idx >> 8;
  int nb  = (int)(f % NBLK);
  int kb  = (int)((f / NBLK) & 7u);
  int tap = (int)(f / ((size_t)NBLK*8));
  int ky = tap/3, kx = tap - ky*3;
  int k0 = kb*32 + (lane >> 4)*16 + 2*j;
  int co = nb*16 + (lane & 15);
  float w0 = w[(((size_t)co*CINCH + k0    )*3 + ky)*3 + kx];
  float w1 = w[(((size_t)co*CINCH + k0 + 1)*3 + ky)*3 + kx];
  unsigned int d = (unsigned int)f2bf(w0) | ((unsigned int)f2bf(w1) << 16);
  wp[idx] = d;
}

// ---------------------------------------------------------------------------
// Implicit-GEMM conv3x3 SAME via v_wmma_f32_16x16x32_bf16.
//   in : NHWC bf16 [B*HW, 256]
//   wp : pre-packed B fragments
//   BF16RELU=1: out = bf16 NHWC, ReLU(x + bias)   (tower layers)
//   BF16RELU=0: out = fp32 [B*HW, CO], x + bias   (logits head)
// Block = 256 threads = 8 waves; each wave computes a 32(M) x 64(N) tile
// (2 A fragments, B fragment reused from registers across both M sub-tiles:
//  24 B of cache traffic per WMMA instead of 40).
// Block tile: 2 M-waves x 4 N-groups = 64 x 256.
// ---------------------------------------------------------------------------
template<int BF16RELU>
__global__ __launch_bounds__(256) void conv_wmma(
    const unsigned short* __restrict__ in,
    const unsigned int*  __restrict__ wp,
    const float*         __restrict__ bias,
    unsigned short*      __restrict__ outB,
    float*               __restrict__ outF,
    int H, int W, int CO)
{
  const int HW = H*W;
  const int M  = BATCHN*HW;
  const int NBLK = CO >> 4;
  const int lane = threadIdx.x & 31;
  const int wave = threadIdx.x >> 5;
  const int wm = wave & 1;            // 2 M-waves per block (32 rows each)
  const int wg = wave >> 1;           // 4 N-groups of 64
  const int mBase = blockIdx.x*64 + wm*32;
  const int nb0 = wg*4;
  int nbCount = NBLK - nb0;
  nbCount = nbCount < 0 ? 0 : (nbCount > 4 ? 4 : nbCount);

  const int half = lane >> 4;
  // two A-fragment row sets: rows mBase+[0,16) and mBase+[16,32)
  const int mg0 = mBase + (lane & 15);
  const int mg1 = mg0 + 16;
  const bool mv0 = mg0 < M;
  const bool mv1 = mg1 < M;
  int b0 = 0, y0 = 0, x0 = 0, b1 = 0, y1 = 0, x1 = 0;
  if (mv0){ b0 = mg0 / HW; int p = mg0 - b0*HW; y0 = p / W; x0 = p - y0*W; }
  if (mv1){ b1 = mg1 / HW; int p = mg1 - b1*HW; y1 = p / W; x1 = p - y1*W; }

  v8f acc0[4] = {};
  v8f acc1[4] = {};
  const v4u vz = {0u,0u,0u,0u};

  if (nbCount > 0){
    for (int tap = 0; tap < 9; ++tap){
      const int ky = tap/3;
      const int dy = ky - 1, dx = (tap - ky*3) - 1;
      const int iy0 = y0 + dy, ix0 = x0 + dx;
      const int iy1 = y1 + dy, ix1 = x1 + dx;
      const bool v0ok = mv0 && iy0 >= 0 && iy0 < H && ix0 >= 0 && ix0 < W;
      const bool v1ok = mv1 && iy1 >= 0 && iy1 < H && ix1 >= 0 && ix1 < W;
      const unsigned short* arow0 = in + (size_t)(b0*HW + iy0*W + ix0)*CINCH;
      const unsigned short* arow1 = in + (size_t)(b1*HW + iy1*W + ix1)*CINCH;
      for (int kb = 0; kb < 8; ++kb){
        const int c0 = kb*32 + half*16;
        Frag a0, a1;
        if (v0ok){
          a0.u[0] = *(const v4u*)(arow0 + c0);       // K[half*16 .. +8)
          a0.u[1] = *(const v4u*)(arow0 + c0 + 8);   // K[half*16+8 .. +8)
        } else { a0.u[0] = vz; a0.u[1] = vz; }
        if (v1ok){
          a1.u[0] = *(const v4u*)(arow1 + c0);
          a1.u[1] = *(const v4u*)(arow1 + c0 + 8);
        } else { a1.u[0] = vz; a1.u[1] = vz; }
        const unsigned int* wb =
            wp + (size_t)((tap*8 + kb)*NBLK)*256 + lane*8;
        #pragma unroll
        for (int q = 0; q < 4; ++q){
          if (q < nbCount){
            Frag bb;
            const unsigned int* wf = wb + (size_t)(nb0 + q)*256;
            bb.u[0] = *(const v4u*)(wf);
            bb.u[1] = *(const v4u*)(wf + 4);
            acc0[q] = __builtin_amdgcn_wmma_f32_16x16x32_bf16(
                false, a0.v, false, bb.v, (short)0, acc0[q], false, false);
            acc1[q] = __builtin_amdgcn_wmma_f32_16x16x32_bf16(
                false, a1.v, false, bb.v, (short)0, acc1[q], false, false);
          }
        }
      }
    }
  }

  // C layout: lane holds column n = lane&15, rows M = r + 8*(lane>>4)
  const int n0 = nb0*16 + (lane & 15);
  #pragma unroll
  for (int t = 0; t < 2; ++t){
    const int ms = mBase + t*16 + half*8;
    #pragma unroll
    for (int q = 0; q < 4; ++q){
      if (q < nbCount){
        const int co = n0 + q*16;
        const float bs = bias[co];
        #pragma unroll
        for (int r = 0; r < 8; ++r){
          const int mm = ms + r;
          if (mm < M){
            float v = (t == 0 ? acc0[q][r] : acc1[q][r]) + bs;
            if (BF16RELU){
              v = v > 0.f ? v : 0.f;
              outB[(size_t)mm*CO + co] = f2bf(v);
            } else {
              outF[(size_t)mm*CO + co] = v;
            }
          }
        }
      }
    }
  }
}

// ---------------------------------------------------------------------------
// Small heads: pred (4 ch, *scale, ReLU) + ctr (1 ch) direct conv3x3.
// One thread per (pixel, co in 0..4).
// ---------------------------------------------------------------------------
__global__ __launch_bounds__(256) void head_small(
    const unsigned short* __restrict__ tow,
    const float* __restrict__ pred_w, const float* __restrict__ pred_b,
    const float* __restrict__ ctr_w,  const float* __restrict__ ctr_b,
    const float* __restrict__ scales, int level,
    float* __restrict__ regOut, float* __restrict__ ctrOut,
    int H, int W)
{
  const int HW = H*W, M = BATCHN*HW;
  int idx = blockIdx.x*256 + threadIdx.x;
  if (idx >= M*5) return;
  int co = idx % 5, mg = idx / 5;
  int b = mg / HW, p2 = mg - b*HW, y = p2 / W, x = p2 - y*W;
  const float* wsel = (co < 4) ? (pred_w + (size_t)co*CINCH*9) : ctr_w;
  float acc = (co < 4) ? pred_b[co] : ctr_b[0];
  for (int ky = 0; ky < 3; ++ky){
    int iy = y + ky - 1; if (iy < 0 || iy >= H) continue;
    for (int kx = 0; kx < 3; ++kx){
      int ix = x + kx - 1; if (ix < 0 || ix >= W) continue;
      const unsigned short* row = tow + (size_t)(b*HW + iy*W + ix)*CINCH;
      const float* wk = wsel + (size_t)ky*3 + kx;
      #pragma unroll 4
      for (int c = 0; c < CINCH; ++c)
        acc += bf2f(row[c]) * wk[(size_t)c*9];
    }
  }
  if (co < 4){
    float v = acc * scales[level];
    regOut[(size_t)mg*4 + co] = v > 0.f ? v : 0.f;
  } else {
    ctrOut[mg] = acc;
  }
}

// ---------------------------------------------------------------------------
// Candidate extraction (deterministic two-pass compaction, no atomics).
// ---------------------------------------------------------------------------
__global__ __launch_bounds__(256) void count_cand(
    const float* __restrict__ logits, const float* __restrict__ ctr,
    int HW, int* __restrict__ counts, int nElem)
{
  const int b = blockIdx.y;
  const int e = blockIdx.x*256 + threadIdx.x;
  int flag = 0;
  if (e < nElem){
    int p = e / NUM_CLASSES, c = e - p*NUM_CLASSES;
    float s = sigmoidf(logits[((size_t)b*HW + p)*NUM_CLASSES + c]) *
              sigmoidf(ctr[(size_t)b*HW + p]);
    flag = (s > PRE_T) ? 1 : 0;
  }
  __shared__ int sh[256];
  sh[threadIdx.x] = flag; __syncthreads();
  for (int s = 128; s > 0; s >>= 1){
    if (threadIdx.x < s) sh[threadIdx.x] += sh[threadIdx.x + s];
    __syncthreads();
  }
  if (threadIdx.x == 0) counts[(size_t)b*gridDim.x + blockIdx.x] = sh[0];
}

__global__ void scan_counts(const int* __restrict__ counts,
                            int* __restrict__ offs, int* __restrict__ totals,
                            int nblk)
{
  const int b = blockIdx.x;
  if (threadIdx.x == 0){
    int acc = 0;
    for (int i = 0; i < nblk; ++i){
      offs[(size_t)b*nblk + i] = acc;
      acc += counts[(size_t)b*nblk + i];
    }
    totals[b] = acc;
  }
}

__global__ __launch_bounds__(256) void write_cand(
    const float* __restrict__ logits, const float* __restrict__ ctr,
    const float* __restrict__ reg, const int* __restrict__ offs,
    const int* __restrict__ cbase, int HW, int W, int stride, int nElem,
    float* __restrict__ cbox, float* __restrict__ cboxO,
    float* __restrict__ cscore, float* __restrict__ carea,
    int* __restrict__ ccls)
{
  const int b = blockIdx.y;
  const int e = blockIdx.x*256 + threadIdx.x;
  int flag = 0; float s = 0.f; int p = 0, c = 0;
  if (e < nElem){
    p = e / NUM_CLASSES; c = e - p*NUM_CLASSES;
    s = sigmoidf(logits[((size_t)b*HW + p)*NUM_CLASSES + c]) *
        sigmoidf(ctr[(size_t)b*HW + p]);
    flag = (s > PRE_T) ? 1 : 0;
  }
  __shared__ int sh[256];
  sh[threadIdx.x] = flag; __syncthreads();
  for (int d = 1; d < 256; d <<= 1){            // inclusive scan
    int v = sh[threadIdx.x];
    int add = (threadIdx.x >= d) ? sh[threadIdx.x - d] : 0;
    __syncthreads();
    sh[threadIdx.x] = v + add;
    __syncthreads();
  }
  if (flag){
    int pos = cbase[b] + offs[(size_t)b*gridDim.x + blockIdx.x] + (sh[threadIdx.x] - 1);
    if (pos < MAXCAND){
      int y = p / W, x = p - y*W;
      float lx = (float)x*(float)stride + (float)(stride >> 1);
      float ly = (float)y*(float)stride + (float)(stride >> 1);
      const float* r = reg + ((size_t)b*HW + p)*4;
      float x1 = lx - r[0], y1 = ly - r[1], x2 = lx + r[2], y2 = ly + r[3];
      float off = (float)c * 10000.0f;
      size_t o = (size_t)b*MAXCAND + pos;
      cbox [o*4+0] = x1;      cbox [o*4+1] = y1;
      cbox [o*4+2] = x2;      cbox [o*4+3] = y2;
      cboxO[o*4+0] = x1+off;  cboxO[o*4+1] = y1+off;
      cboxO[o*4+2] = x2+off;  cboxO[o*4+3] = y2+off;
      carea[o]  = (x2 - x1)*(y2 - y1);
      cscore[o] = sqrtf(fmaxf(s, 1e-12f));
      ccls[o]   = c;
    }
  }
}

__global__ void bump_base(const int* __restrict__ totals, int* __restrict__ cbase){
  int b = threadIdx.x;
  if (b < BATCHN){
    int nb = cbase[b] + totals[b];
    cbase[b] = nb > MAXCAND ? MAXCAND : nb;
  }
}

__global__ void init_base(int* __restrict__ cbase){
  if (threadIdx.x < BATCHN) cbase[threadIdx.x] = 0;
}

// ---------------------------------------------------------------------------
// Greedy class-aware NMS (one block per batch image).
// ---------------------------------------------------------------------------
__global__ __launch_bounds__(256) void nms_kernel(
    const float* __restrict__ cbox, const float* __restrict__ cboxO,
    float* __restrict__ cscore, const float* __restrict__ carea,
    const int* __restrict__ ccls, const int* __restrict__ cbase,
    float* __restrict__ out)
{
  const int b = blockIdx.x;
  int n = cbase[b]; if (n > MAXCAND) n = MAXCAND;
  const float* box  = cbox  + (size_t)b*MAXCAND*4;
  const float* boxO = cboxO + (size_t)b*MAXCAND*4;
  float*       sc   = cscore + (size_t)b*MAXCAND;
  const float* ar   = carea + (size_t)b*MAXCAND;
  const int*   cl   = ccls  + (size_t)b*MAXCAND;
  __shared__ float sv[256];
  __shared__ int   si[256];

  for (int it = 0; it < POST_K; ++it){
    float best = -1e30f; int bi = -1;
    for (int i = threadIdx.x; i < n; i += 256){
      float v = sc[i];
      if (v > best){ best = v; bi = i; }
    }
    sv[threadIdx.x] = best; si[threadIdx.x] = bi;
    __syncthreads();
    for (int s = 128; s > 0; s >>= 1){
      if (threadIdx.x < s){
        float v2 = sv[threadIdx.x + s]; int i2 = si[threadIdx.x + s];
        float v1 = sv[threadIdx.x];     int i1 = si[threadIdx.x];
        if (v2 > v1 || (v2 == v1 && i2 >= 0 && (i1 < 0 || i2 < i1))){
          sv[threadIdx.x] = v2; si[threadIdx.x] = i2;
        }
      }
      __syncthreads();
    }
    int sel = si[0]; float val = sv[0];
    if (threadIdx.x == 0){
      float* o = out + ((size_t)b*POST_K + it)*6;
      if (sel >= 0 && val > 0.f){
        o[0] = box[(size_t)sel*4+0]; o[1] = box[(size_t)sel*4+1];
        o[2] = box[(size_t)sel*4+2]; o[3] = box[(size_t)sel*4+3];
        o[4] = val; o[5] = (float)cl[sel];
      } else {
        o[0]=0.f;o[1]=0.f;o[2]=0.f;o[3]=0.f;o[4]=0.f;o[5]=0.f;
      }
    }
    __syncthreads();
    if (sel >= 0){
      float x1 = boxO[(size_t)sel*4+0], y1 = boxO[(size_t)sel*4+1];
      float x2 = boxO[(size_t)sel*4+2], y2 = boxO[(size_t)sel*4+3];
      float a0 = ar[sel];
      for (int i = threadIdx.x; i < n; i += 256){
        float xx1 = fmaxf(x1, boxO[(size_t)i*4+0]);
        float yy1 = fmaxf(y1, boxO[(size_t)i*4+1]);
        float xx2 = fminf(x2, boxO[(size_t)i*4+2]);
        float yy2 = fminf(y2, boxO[(size_t)i*4+3]);
        float w = fmaxf(xx2 - xx1, 0.f), h = fmaxf(yy2 - yy1, 0.f);
        float inter = w*h;
        float iou = inter / (a0 + ar[i] - inter + 1e-9f);
        if (iou > NMS_T) sc[i] = -1.f;   // includes self (iou==1)
      }
    }
    __syncthreads();
  }
}

// ---------------------------------------------------------------------------
extern "C" void kernel_launch(void* const* d_in, const int* in_sizes, int n_in,
                              void* d_out, int out_size, void* d_ws, size_t ws_size,
                              hipStream_t stream)
{
  const float* feats[5] = {(const float*)d_in[0], (const float*)d_in[1],
                           (const float*)d_in[2], (const float*)d_in[3],
                           (const float*)d_in[4]};
  const float* cls_w   = (const float*)d_in[5];
  const float* cls_b   = (const float*)d_in[6];
  const float* box_w   = (const float*)d_in[7];
  const float* box_b   = (const float*)d_in[8];
  const float* logit_w = (const float*)d_in[9];
  const float* logit_b = (const float*)d_in[10];
  const float* pred_w  = (const float*)d_in[11];
  const float* pred_b  = (const float*)d_in[12];
  const float* ctr_w   = (const float*)d_in[13];
  const float* ctr_b   = (const float*)d_in[14];
  const float* scales  = (const float*)d_in[15];

  // ---- carve workspace -------------------------------------------------
  char* wsp = (char*)d_ws;
  auto alloc = [&](size_t bytes)->char*{
    char* r = wsp; wsp += (bytes + 255) & ~(size_t)255; return r;
  };
  const size_t packTowerBytes = (size_t)9*8*16*256*4;   // 1,179,648 B
  const size_t packLogitBytes = (size_t)9*8*5*256*4;
  unsigned int* packT[8];
  for (int i = 0; i < 8; ++i) packT[i] = (unsigned int*)alloc(packTowerBytes);
  unsigned int* packL = (unsigned int*)alloc(packLogitBytes);

  const int HWmax = 100*152;
  unsigned short* X0 = (unsigned short*)alloc((size_t)BATCHN*HWmax*CINCH*2);
  unsigned short* T1 = (unsigned short*)alloc((size_t)BATCHN*HWmax*CINCH*2);
  unsigned short* T2 = (unsigned short*)alloc((size_t)BATCHN*HWmax*CINCH*2);
  float* Lbuf = (float*)alloc((size_t)BATCHN*HWmax*NUM_CLASSES*4);
  float* ctrB = (float*)alloc((size_t)BATCHN*HWmax*4);
  float* regB = (float*)alloc((size_t)BATCHN*HWmax*4*4);
  const int BLKMAX = (HWmax*NUM_CLASSES + 255)/256;
  int* counts = (int*)alloc((size_t)BATCHN*BLKMAX*4);
  int* offs   = (int*)alloc((size_t)BATCHN*BLKMAX*4);
  int* totals = (int*)alloc(BATCHN*sizeof(int));
  float* cbox  = (float*)alloc((size_t)BATCHN*MAXCAND*4*4);
  float* cboxO = (float*)alloc((size_t)BATCHN*MAXCAND*4*4);
  float* cscore= (float*)alloc((size_t)BATCHN*MAXCAND*4);
  float* carea = (float*)alloc((size_t)BATCHN*MAXCAND*4);
  int*   ccls  = (int*)alloc((size_t)BATCHN*MAXCAND*4);
  int*   cbase = (int*)alloc(64);
  (void)ws_size; (void)n_in; (void)in_sizes; (void)out_size;

  // ---- pack weights once per launch ------------------------------------
  {
    size_t nd = (size_t)9*8*16*256;
    dim3 g((unsigned)((nd + 255)/256));
    for (int i = 0; i < 4; ++i)
      pack_weights<<<g,256,0,stream>>>(cls_w + (size_t)i*CINCH*CINCH*9, packT[i], 256);
    for (int i = 0; i < 4; ++i)
      pack_weights<<<g,256,0,stream>>>(box_w + (size_t)i*CINCH*CINCH*9, packT[4+i], 256);
  }
  {
    size_t nd = (size_t)9*8*5*256;
    dim3 g((unsigned)((nd + 255)/256));
    pack_weights<<<g,256,0,stream>>>(logit_w, packL, 80);
  }
  init_base<<<1,32,0,stream>>>(cbase);

  const int Hs[5] = {100,50,25,13,7};
  const int Ws[5] = {152,76,38,19,10};
  const int Ss[5] = {8,16,32,64,128};

  for (int l = 0; l < 5; ++l){
    const int H = Hs[l], W = Ws[l], HW = H*W, M = BATCHN*HW;
    {
      size_t tot = (size_t)M*CINCH;
      dim3 g((unsigned)((tot + 255)/256));
      to_nhwc<<<g,256,0,stream>>>(feats[l], X0, HW);
    }
    dim3 gc((unsigned)((M + 63)/64));
    // cls tower: X0 -> T1 -> T2 -> T1 -> T2
    conv_wmma<1><<<gc,256,0,stream>>>(X0, packT[0], cls_b +   0, T1, nullptr, H, W, 256);
    conv_wmma<1><<<gc,256,0,stream>>>(T1, packT[1], cls_b + 256, T2, nullptr, H, W, 256);
    conv_wmma<1><<<gc,256,0,stream>>>(T2, packT[2], cls_b + 512, T1, nullptr, H, W, 256);
    conv_wmma<1><<<gc,256,0,stream>>>(T1, packT[3], cls_b + 768, T2, nullptr, H, W, 256);
    // logits head (fp32 out)
    conv_wmma<0><<<gc,256,0,stream>>>(T2, packL, logit_b, nullptr, Lbuf, H, W, 80);
    // box tower: X0 -> T1 -> T2 -> T1 -> T2 (stream-ordered after logits)
    conv_wmma<1><<<gc,256,0,stream>>>(X0, packT[4], box_b +   0, T1, nullptr, H, W, 256);
    conv_wmma<1><<<gc,256,0,stream>>>(T1, packT[5], box_b + 256, T2, nullptr, H, W, 256);
    conv_wmma<1><<<gc,256,0,stream>>>(T2, packT[6], box_b + 512, T1, nullptr, H, W, 256);
    conv_wmma<1><<<gc,256,0,stream>>>(T1, packT[7], box_b + 768, T2, nullptr, H, W, 256);
    // ctr + pred heads
    {
      size_t tot = (size_t)M*5;
      dim3 g((unsigned)((tot + 255)/256));
      head_small<<<g,256,0,stream>>>(T2, pred_w, pred_b, ctr_w, ctr_b,
                                     scales, l, regB, ctrB, H, W);
    }
    // deterministic candidate compaction
    const int nElem = HW*NUM_CLASSES;
    const int nblk  = (nElem + 255)/256;
    count_cand<<<dim3((unsigned)nblk, BATCHN),256,0,stream>>>(Lbuf, ctrB, HW, counts, nElem);
    scan_counts<<<BATCHN,32,0,stream>>>(counts, offs, totals, nblk);
    write_cand<<<dim3((unsigned)nblk, BATCHN),256,0,stream>>>(
        Lbuf, ctrB, regB, offs, cbase, HW, W, Ss[l], nElem,
        cbox, cboxO, cscore, carea, ccls);
    bump_base<<<1,32,0,stream>>>(totals, cbase);
  }

  nms_kernel<<<BATCHN,256,0,stream>>>(cbox, cboxO, cscore, carea, ccls, cbase,
                                      (float*)d_out);
}
